// FNNModel_85899345920311
// MI455X (gfx1250) — matrix operand order repaired and checked by reading, not compiled, not run on hardware
//
#include <hip/hip_runtime.h>
#include <hip/hip_bf16.h>
#include <math.h>

typedef __attribute__((ext_vector_type(16))) _Float16 v16h;
typedef __attribute__((ext_vector_type(8)))  _Float16 v8h;
typedef __attribute__((ext_vector_type(8)))  float    v8f;

#define NTOKEN 50257
#define EMB    128
#define WINSZ  5
#define HID    256
#define SEQ    256
#define BATCH  8
#define LOUT   (SEQ - WINSZ + 1)   // 252
#define NROW   (LOUT * BATCH)      // 2016  (multiple of 16)
#define KDIM   HID                 // 256   (8 x 32 K-steps)
#define NCG    393                 // column groups of 128 cols (393*128 = 50304 >= 50257)
#define VPAD   (NCG * 128)         // 50304

// ---------------------------------------------------------------- dec_w -> f16
__global__ void k_convert_w(const float* __restrict__ dec_w,
                            _Float16* __restrict__ w16) {
    int n = blockIdx.x;          // padded vocab row
    int k = threadIdx.x;         // hidden index (256 threads)
    float v = (n < NTOKEN) ? dec_w[(size_t)n * HID + k] : 0.0f;
    w16[(size_t)n * HID + k] = (_Float16)v;
}

// ------------------------------------------- embedding gather + mid GEMM + tanh
__global__ void k_embed_mid(const int* __restrict__ tokens,
                            const float* __restrict__ enc_w,
                            const float* __restrict__ mid_w,
                            const float* __restrict__ mid_b,
                            _Float16* __restrict__ h16) {
    __shared__ __align__(16) float win[WINSZ * EMB];   // 640 floats
    const int row = blockIdx.x;          // 0..2015
    const int lo  = row / BATCH;
    const int b   = row % BATCH;
    for (int t = threadIdx.x; t < WINSZ * EMB; t += blockDim.x) {
        int w = t >> 7, e = t & 127;
        int tok = tokens[(lo + w) * BATCH + b];
        win[t] = enc_w[(size_t)tok * EMB + e];
    }
    __syncthreads();
    const int j = threadIdx.x;           // hidden unit, 256 threads
    float acc = mid_b[j];
    const float4* mw = (const float4*)(mid_w + (size_t)j * (WINSZ * EMB));
    const float4* wv = (const float4*)win;
    #pragma unroll 4
    for (int k4 = 0; k4 < (WINSZ * EMB) / 4; ++k4) {
        float4 a = mw[k4];
        float4 c = wv[k4];
        acc += a.x * c.x + a.y * c.y + a.z * c.z + a.w * c.w;
    }
    h16[(size_t)row * HID + j] = (_Float16)tanhf(acc);
}

// --------------- decoder GEMM (WMMA f16 -> f32) + bias + online softmax partials
__global__ void k_decoder_gemm(const _Float16* __restrict__ h16,
                               const _Float16* __restrict__ w16,
                               const float* __restrict__ dec_b,
                               float* __restrict__ out,
                               float2* __restrict__ partials) {
    __shared__ float lds_m[8][16];
    __shared__ float lds_s[8][16];

    const int lane = threadIdx.x & 31;
    const int wave = threadIdx.x >> 5;        // 8 waves per WG
    const int row0 = blockIdx.y * 16;         // 16-row tile
    const int cg   = blockIdx.x;              // 128-col group
    const int n0   = (cg * 8 + wave) * 16;    // this wave's 16-col tile
    const int m    = lane & 15;
    const int hi   = lane >> 4;               // half-wave select
    const int n    = n0 + m;                  // padded column index (< VPAD)

    const _Float16* ha = h16 + (size_t)(row0 + m) * KDIM;   // A: row m of tile
    const _Float16* wb = w16 + (size_t)n * KDIM;            // B: column n

    v8f c = {};
    #pragma unroll
    for (int kk = 0; kk < KDIM; kk += 32) {
        v16h a, bm;
        // A 16x32 f16 layout: lane<16 -> K = kk+{0..7,16..23}; lane>=16 -> +8
        ((v8h*)&a)[0]  = *(const v8h*)(ha + kk + hi * 8);
        ((v8h*)&a)[1]  = *(const v8h*)(ha + kk + 16 + hi * 8);
        // B 32x16 f16 layout: lane<16 -> K = kk+{0..7,16..23}; lane>=16 -> +8
        ((v8h*)&bm)[0] = *(const v8h*)(wb + kk + hi * 8);
        ((v8h*)&bm)[1] = *(const v8h*)(wb + kk + 16 + hi * 8);
        c = __builtin_amdgcn_wmma_f32_16x16x32_f16(false, a, false, bm,
                                                   (short)0, c, false, false);
    }

    const bool  valid = (n < NTOKEN);
    const float bias  = valid ? dec_b[n] : 0.0f;

    float mloc[8], sloc[8];
    #pragma unroll
    for (int r = 0; r < 8; ++r) {
        // C/D layout: VGPR r, lanes 0-15 -> row r; lanes 16-31 -> row r+8
        float logit = c[r] + bias;
        int row = row0 + r + hi * 8;
        if (valid) out[(size_t)row * NTOKEN + n] = logit;
        // per-row max over this 16-col tile (butterfly within 16-lane group)
        float mv = valid ? logit : -INFINITY;
        #pragma unroll
        for (int off = 1; off < 16; off <<= 1)
            mv = fmaxf(mv, __shfl_xor(mv, off, 32));
        float ev = valid ? __expf(logit - mv) : 0.0f;
        #pragma unroll
        for (int off = 1; off < 16; off <<= 1)
            ev += __shfl_xor(ev, off, 32);
        mloc[r] = mv;
        sloc[r] = ev;
    }
    if (m == 0) {
        #pragma unroll
        for (int r = 0; r < 8; ++r) {
            lds_m[wave][r + hi * 8] = mloc[r];
            lds_s[wave][r + hi * 8] = sloc[r];
        }
    }
    __syncthreads();
    // merge 8 wave tiles (128 cols) -> one (max, sumexp) partial per row
    if (threadIdx.x < 16) {
        int t = threadIdx.x;
        float M = -INFINITY;
        #pragma unroll
        for (int w = 0; w < 8; ++w) M = fmaxf(M, lds_m[w][t]);
        float S = 0.0f;
        #pragma unroll
        for (int w = 0; w < 8; ++w) {
            float sw = lds_s[w][t];
            if (sw > 0.0f) S += sw * __expf(lds_m[w][t] - M);
        }
        partials[(size_t)(row0 + t) * NCG + cg] = make_float2(M, S);
    }
}

// ------------------------------------------------ per-row merge of 393 partials
__global__ void k_rowstats(const float2* __restrict__ partials,
                           float* __restrict__ rowM,
                           float* __restrict__ rowLS) {
    int row = blockIdx.x * blockDim.x + threadIdx.x;
    if (row >= NROW) return;
    const float2* p = partials + (size_t)row * NCG;
    float M = -INFINITY;
    for (int cg = 0; cg < NCG; ++cg) M = fmaxf(M, p[cg].x);
    float S = 0.0f;
    for (int cg = 0; cg < NCG; ++cg) {
        float2 q = p[cg];
        if (q.y > 0.0f) S += q.y * __expf(q.x - M);
    }
    rowM[row]  = M;
    rowLS[row] = logf(S);
}

// ------------------------------------------------- out = logit - max - log(sum)
__global__ void k_finalize(float* __restrict__ out,
                           const float* __restrict__ rowM,
                           const float* __restrict__ rowLS) {
    int row = blockIdx.y;
    int col = blockIdx.x * blockDim.x + threadIdx.x;
    if (col < NTOKEN) {
        float sub = rowM[row] + rowLS[row];
        size_t i = (size_t)row * NTOKEN + col;
        out[i] = out[i] - sub;
    }
}

extern "C" void kernel_launch(void* const* d_in, const int* in_sizes, int n_in,
                              void* d_out, int out_size, void* d_ws, size_t ws_size,
                              hipStream_t stream) {
    const int*   tokens = (const int*)  d_in[0];
    const float* enc_w  = (const float*)d_in[1];
    const float* mid_w  = (const float*)d_in[2];
    const float* mid_b  = (const float*)d_in[3];
    const float* dec_w  = (const float*)d_in[4];
    const float* dec_b  = (const float*)d_in[5];
    float* out = (float*)d_out;

    // workspace carve-out (~33 MB total)
    char* ws = (char*)d_ws;
    size_t off = 0;
    _Float16* w16 = (_Float16*)(ws + off);
    off += (size_t)VPAD * HID * sizeof(_Float16);  off = (off + 255) & ~(size_t)255;
    _Float16* h16 = (_Float16*)(ws + off);
    off += (size_t)NROW * HID * sizeof(_Float16);  off = (off + 255) & ~(size_t)255;
    float2* partials = (float2*)(ws + off);
    off += (size_t)NROW * NCG * sizeof(float2);    off = (off + 255) & ~(size_t)255;
    float* rowM = (float*)(ws + off);  off += (size_t)NROW * sizeof(float);
    float* rowLS = (float*)(ws + off); off += (size_t)NROW * sizeof(float);

    k_convert_w<<<VPAD, HID, 0, stream>>>(dec_w, w16);
    k_embed_mid<<<NROW, HID, 0, stream>>>(tokens, enc_w, mid_w, mid_b, h16);

    dim3 gg(NCG, NROW / 16);
    k_decoder_gemm<<<gg, 256, 0, stream>>>(h16, w16, dec_b, out, partials);

    k_rowstats<<<(NROW + 255) / 256, 256, 0, stream>>>(partials, rowM, rowLS);

    dim3 gf((NTOKEN + 255) / 256, NROW);
    k_finalize<<<gf, 256, 0, stream>>>(out, rowM, rowLS);
}
